// QuantizedLinear_28724741276095
// MI455X (gfx1250) — compile-verified, matrix operand order checked by reading, not run on hardware
//
#include <hip/hip_runtime.h>
#include <hip/hip_bf16.h>

typedef __attribute__((ext_vector_type(16))) __bf16 v16bf;
typedef __attribute__((ext_vector_type(8)))  float  v8f;
typedef __attribute__((ext_vector_type(4)))  int    v4i;

#define N_FEAT 4096
#define ROWS   8192
#define RANK   64

// ---------------------------------------------------------------------------
// CDNA5 async global->LDS copy (ASYNCcnt path), with safe fallback
// ---------------------------------------------------------------------------
#if defined(__gfx1250__) && __has_builtin(__builtin_amdgcn_global_load_async_to_lds_b128)
#define HAVE_ASYNC_LDS 1
#else
#define HAVE_ASYNC_LDS 0
#endif

#if HAVE_ASYNC_LDS
typedef __attribute__((address_space(1))) v4i* glb_v4i_ptr;
typedef __attribute__((address_space(3))) v4i* lds_v4i_ptr;
#endif

__device__ __forceinline__ void cp16_g2l(__bf16* lds, const __bf16* g) {
#if HAVE_ASYNC_LDS
    __builtin_amdgcn_global_load_async_to_lds_b128(
        (glb_v4i_ptr)(uintptr_t)g,
        (lds_v4i_ptr)(unsigned)(uintptr_t)lds,
        0, 0);
#else
    *(uint4*)lds = *(const uint4*)g;
#endif
}

__device__ __forceinline__ void async_copy_wait() {
#if HAVE_ASYNC_LDS
#if __has_builtin(__builtin_amdgcn_s_wait_asynccnt)
    __builtin_amdgcn_s_wait_asynccnt(0);
#else
    asm volatile("s_wait_asynccnt 0" ::: "memory");
#endif
#endif
}

// ---------------------------------------------------------------------------
// FWHT of 4096 floats in LDS, 256 threads, 12 stages
// ---------------------------------------------------------------------------
__device__ __forceinline__ void fwht4096(float* xs, int t) {
    for (int h = 1; h < N_FEAT; h <<= 1) {
        __syncthreads();
        #pragma unroll 1
        for (int p = t; p < N_FEAT / 2; p += 256) {
            int i = ((p & ~(h - 1)) << 1) | (p & (h - 1));
            int j = i + h;
            float a = xs[i], b = xs[j];
            xs[i] = a + b;
            xs[j] = a - b;
        }
    }
    __syncthreads();
}

// ---------------------------------------------------------------------------
// Kernel 1: x = (input/scaleWH*SU); x = FWHT(x)/64; write bf16 xh; xB = x@B^T
// ---------------------------------------------------------------------------
__global__ __launch_bounds__(256) void prep_kernel(
    const float* __restrict__ in, const float* __restrict__ scaleWH,
    const float* __restrict__ SU, const float* __restrict__ Bm,
    __bf16* __restrict__ xh, float* __restrict__ xB)
{
    __shared__ float xs[N_FEAT];
    __shared__ float xb[RANK];
    const int row = blockIdx.x;
    const int t   = threadIdx.x;

    for (int i = t; i < N_FEAT; i += 256)
        xs[i] = in[(size_t)row * N_FEAT + i] / scaleWH[i] * SU[i];
    if (t < RANK) xb[t] = 0.0f;

    fwht4096(xs, t);

    const float inv = 1.0f / 64.0f;   // 1/sqrt(4096)
    for (int i = t; i < N_FEAT; i += 256)
        xh[(size_t)row * N_FEAT + i] = (__bf16)(xs[i] * inv);

    // rank-64 projection: xB[row] = x @ B^T  (B is RANK x N_FEAT row-major)
    float loc[16];
    const int k0 = t * 16;
    #pragma unroll
    for (int i = 0; i < 16; ++i) loc[i] = xs[k0 + i] * inv;
    #pragma unroll 1
    for (int j = 0; j < RANK; ++j) {
        float p = 0.0f;
        #pragma unroll
        for (int i = 0; i < 16; ++i) p += loc[i] * Bm[(size_t)j * N_FEAT + k0 + i];
        atomicAdd(&xb[j], p);
    }
    __syncthreads();
    if (t < RANK) xB[(size_t)row * RANK + t] = xb[t];
}

// ---------------------------------------------------------------------------
// Kernel 2: decode W to bf16 (one 8-wide code group per thread)
// ---------------------------------------------------------------------------
__global__ __launch_bounds__(256) void decode_kernel(
    const int* __restrict__ Qidxs, const float* __restrict__ cb,
    __bf16* __restrict__ wh)
{
    const size_t g = (size_t)blockIdx.x * 256 + threadIdx.x;   // 0 .. 2M-1
    const int code = Qidxs[g];
    const float4* src = (const float4*)(cb + (size_t)code * 8);
    float4 c0 = src[0], c1 = src[1];
    union { __bf16 h[8]; uint4 u; } o;
    o.h[0] = (__bf16)c0.x; o.h[1] = (__bf16)c0.y;
    o.h[2] = (__bf16)c0.z; o.h[3] = (__bf16)c0.w;
    o.h[4] = (__bf16)c1.x; o.h[5] = (__bf16)c1.y;
    o.h[6] = (__bf16)c1.z; o.h[7] = (__bf16)c1.w;
    *(uint4*)(wh + g * 8) = o.u;
}

// ---------------------------------------------------------------------------
// Kernel 3: z = xh @ Wh^T   (M=8192, N=4096, K=4096), bf16 WMMA, f32 accum
// 128x128 tile, BK=64, 256 threads = 8 waves (4 over M x 2 over N),
// double-buffered LDS with async global->LDS prefetch
// ---------------------------------------------------------------------------
#define LDSTR 72   // 64 + 8 bf16 pad: 36-dword row stride -> conflict-free
#define NKSTEP (N_FEAT / 64)

__shared__ alignas(16) __bf16 g_xs[2][128 * LDSTR];
__shared__ alignas(16) __bf16 g_ws[2][128 * LDSTR];

__device__ __forceinline__ void gemm_prefetch(
    int buf, int kb, int tid, int m0, int n0,
    const __bf16* __restrict__ xh, const __bf16* __restrict__ wh)
{
    #pragma unroll
    for (int l = 0; l < 4; ++l) {
        int idx = tid + l * 256;          // 0..1023
        int row = idx >> 3;               // 0..127
        int ch  = (idx & 7) << 3;         // 0,8,..,56
        cp16_g2l(&g_xs[buf][row * LDSTR + ch],
                 xh + (size_t)(m0 + row) * N_FEAT + kb + ch);
        cp16_g2l(&g_ws[buf][row * LDSTR + ch],
                 wh + (size_t)(n0 + row) * N_FEAT + kb + ch);
    }
}

__device__ __forceinline__ void gemm_compute(
    int buf, int wm, int wn, int half, int lr, v8f (&acc)[2][4])
{
    const __bf16* xs = g_xs[buf];
    const __bf16* ws = g_ws[buf];
    #pragma unroll
    for (int kk = 0; kk < 64; kk += 32) {
        union Frag { v16bf v; unsigned u[8]; };
        Frag a[2], b[4];
        // A fragments: 16x32 bf16, lane = (half,lr); VGPR v holds a K pair
        #pragma unroll
        for (int mi = 0; mi < 2; ++mi) {
            int row = wm * 32 + mi * 16 + lr;
            #pragma unroll
            for (int v = 0; v < 8; ++v) {
                int k = kk + (v < 4 ? 2 * v + 8 * half
                                    : 16 + 2 * (v - 4) + 8 * half);
                a[mi].u[v] = *(const unsigned*)(&xs[row * LDSTR + k]);
            }
        }
        // B fragments: 32x16 bf16; B[k][n] = W[n][k] -> read W tile row n
        #pragma unroll
        for (int ni = 0; ni < 4; ++ni) {
            int col = wn * 64 + ni * 16 + lr;
            #pragma unroll
            for (int v = 0; v < 8; ++v) {
                int k = kk + (v < 4 ? 2 * v + 8 * half
                                    : 16 + 2 * (v - 4) + 8 * half);
                b[ni].u[v] = *(const unsigned*)(&ws[col * LDSTR + k]);
            }
        }
        #pragma unroll
        for (int mi = 0; mi < 2; ++mi)
            #pragma unroll
            for (int ni = 0; ni < 4; ++ni)
                acc[mi][ni] = __builtin_amdgcn_wmma_f32_16x16x32_bf16(
                    false, a[mi].v, false, b[ni].v,
                    (short)0, acc[mi][ni], false, false);
    }
}

__global__ __launch_bounds__(256) void gemm_kernel(
    const __bf16* __restrict__ xh, const __bf16* __restrict__ wh,
    float* __restrict__ z)
{
    const int tid  = threadIdx.x;
    const int n0   = blockIdx.x * 128;
    const int m0   = blockIdx.y * 128;
    const int wid  = tid >> 5;
    const int lane = tid & 31;
    const int wm   = wid & 3;      // 4 waves over M, 32 rows each
    const int wn   = wid >> 2;     // 2 waves over N, 64 cols each
    const int half = lane >> 4;
    const int lr   = lane & 15;

    v8f acc[2][4];
    #pragma unroll
    for (int mi = 0; mi < 2; ++mi)
        #pragma unroll
        for (int ni = 0; ni < 4; ++ni) { v8f zv = {}; acc[mi][ni] = zv; }

    gemm_prefetch(0, 0, tid, m0, n0, xh, wh);

    #pragma unroll 1
    for (int kbi = 0; kbi < NKSTEP; ++kbi) {
        const int cur = kbi & 1;
        async_copy_wait();          // my prefetch into buf[cur] has landed
        __syncthreads();            // everyone's landed; buf[1-cur] free
        if (kbi + 1 < NKSTEP)
            gemm_prefetch(1 - cur, (kbi + 1) * 64, tid, m0, n0, xh, wh);
        gemm_compute(cur, wm, wn, half, lr, acc);
    }

    // ---- write back: C/D layout — lane lr = N, VGPR r -> M = r + 8*half
    #pragma unroll
    for (int mi = 0; mi < 2; ++mi) {
        #pragma unroll
        for (int ni = 0; ni < 4; ++ni) {
            int n     = n0 + wn * 64 + ni * 16 + lr;
            int mbase = m0 + wm * 32 + mi * 16 + 8 * half;
            #pragma unroll
            for (int r = 0; r < 8; ++r)
                z[(size_t)(mbase + r) * N_FEAT + n] = acc[mi][ni][r];
        }
    }
}

// ---------------------------------------------------------------------------
// Kernel 4: z' = FWHT(z)/64; out = (z' + xB@A^T)*SV + bias
// ---------------------------------------------------------------------------
__global__ __launch_bounds__(256) void final_kernel(
    const float* __restrict__ z, const float* __restrict__ xB,
    const float* __restrict__ Am, const float* __restrict__ SV,
    const float* __restrict__ bias, float* __restrict__ out)
{
    __shared__ float zs[N_FEAT];
    __shared__ float xb[RANK];
    const int row = blockIdx.x;
    const int t   = threadIdx.x;

    for (int i = t; i < N_FEAT; i += 256)
        zs[i] = z[(size_t)row * N_FEAT + i];
    if (t < RANK) xb[t] = xB[(size_t)row * RANK + t];

    fwht4096(zs, t);

    const float inv = 1.0f / 64.0f;
    const int j0 = t * 16;
    #pragma unroll 1
    for (int i = 0; i < 16; ++i) {
        int j = j0 + i;
        float ab = 0.0f;
        #pragma unroll
        for (int r = 0; r < RANK; ++r) ab += xb[r] * Am[(size_t)j * RANK + r];
        out[(size_t)row * N_FEAT + j] = (zs[j] * inv + ab) * SV[j] + bias[j];
    }
}

// ---------------------------------------------------------------------------
extern "C" void kernel_launch(void* const* d_in, const int* in_sizes, int n_in,
                              void* d_out, int out_size, void* d_ws, size_t ws_size,
                              hipStream_t stream)
{
    const float* input   = (const float*)d_in[0];
    const int*   Qidxs   = (const int*)  d_in[1];
    const float* cb      = (const float*)d_in[2];
    const float* SU      = (const float*)d_in[3];
    const float* SV      = (const float*)d_in[4];
    const float* A       = (const float*)d_in[5];
    const float* B       = (const float*)d_in[6];
    const float* scaleWH = (const float*)d_in[7];
    const float* bias    = (const float*)d_in[8];

    char* ws = (char*)d_ws;
    __bf16* xh = (__bf16*)(ws);                         //  64 MB
    __bf16* wh = (__bf16*)(ws + (64ull  << 20));        //  32 MB
    float*  xB = (float*) (ws + (96ull  << 20));        //   2 MB
    float*  z  = (float*) (ws + (98ull  << 20));        // 128 MB
    float*  out = (float*)d_out;

    prep_kernel  <<<ROWS, 256, 0, stream>>>(input, scaleWH, SU, B, xh, xB);
    decode_kernel<<<(N_FEAT * (N_FEAT / 8)) / 256, 256, 0, stream>>>(Qidxs, cb, wh);
    dim3 g(N_FEAT / 128, ROWS / 128);   // (32, 64)
    gemm_kernel  <<<g, 256, 0, stream>>>(xh, wh, z);
    final_kernel <<<ROWS, 256, 0, stream>>>(z, xB, A, SV, bias, out);
}
